// NeuralCDE_83640193122574
// MI455X (gfx1250) — matrix operand order repaired and checked by reading, not compiled, not run on hardware
//
#include <hip/hip_runtime.h>
#include <hip/hip_bf16.h>

typedef __attribute__((ext_vector_type(8)))  _Float16 v8h;
typedef __attribute__((ext_vector_type(16))) _Float16 v16h;
typedef __attribute__((ext_vector_type(8)))  float    v8f;
typedef __attribute__((ext_vector_type(4)))  unsigned v4u;
typedef __attribute__((ext_vector_type(8)))  int      v8i;
typedef __attribute__((ext_vector_type(4)))  int      v4i;

#define TILE_B 16
#define NH     64
#define ND     32
#define NF     2048
#define NSTEP  1023
#define CHUNK  1024

#if __has_builtin(__builtin_amdgcn_tensor_load_to_lds)
#define HAVE_TDM 1
#endif

// A-matrix fragment (16x32 f16): per lane, two contiguous 8-half runs:
//   halves 0..7  <- K = hi*16     .. hi*16+7
//   halves 8..15 <- K = hi*16+8   .. hi*16+15
__device__ __forceinline__ v16h loadA(const _Float16* row, int hi) {
  v8h lo = *(const v8h*)(row + hi * 16);
  v8h up = *(const v8h*)(row + hi * 16 + 8);
  return __builtin_shufflevector(lo, up, 0, 1, 2, 3, 4, 5, 6, 7,
                                 8, 9, 10, 11, 12, 13, 14, 15);
}

__device__ __forceinline__ v8f wmma16(v16h a, v16h b, v8f c) {
  return __builtin_amdgcn_wmma_f32_16x16x32_f16(false, a, false, b, (short)0, c,
                                                false, false);
}

__device__ __forceinline__ float dev_tanh(float x) {
#if __has_builtin(__builtin_amdgcn_tanhf)
  return __builtin_amdgcn_tanhf(x);
#else
  float e = __expf(2.0f * x);
  return 1.0f - 2.0f / (e + 1.0f);
#endif
}

__global__ __launch_bounds__(256, 1)
void neural_cde_kernel(const float* __restrict__ coeffs, const float* __restrict__ initial,
                       const float* __restrict__ W_init, const float* __restrict__ b_init,
                       const float* __restrict__ W1, const float* __restrict__ b1,
                       const float* __restrict__ W2, const float* __restrict__ b2,
                       const float* __restrict__ W3, const float* __restrict__ b3,
                       const float* __restrict__ Wf, const float* __restrict__ bf,
                       const float* __restrict__ W_out, const float* __restrict__ b_out,
                       float* __restrict__ out) {
  // ---- resident state (per 16-row batch tile) ----
  __shared__ alignas(32) _Float16 sW1t[16 * 64];         // [n][k] k-contiguous B-frag order
  __shared__ alignas(32) _Float16 sW2t[16 * 32];         // [n][k], k>=15 zero
  __shared__ alignas(32) _Float16 sW3t[16 * 32];
  __shared__ alignas(32) _Float16 sWfsw[128 * 16 * 16];  // [tile][n][k] B-frag order
  __shared__ float sbf[NF];
  __shared__ float sb1[16], sb2[16], sb3[16];
  __shared__ float sh[TILE_B * NH];                      // ODE state h (f32)
  __shared__ float sk1[TILE_B * NH], sk2[TILE_B * NH], sk3[TILE_B * NH];
  __shared__ float skc[TILE_B * NH];                     // current k
  __shared__ alignas(32) _Float16 shevh[TILE_B * NH];    // h at eval point (f16, A-layout-friendly)
  __shared__ alignas(32) _Float16 sza[TILE_B * 32];      // z buffers, K padded to 32 (upper half 0)
  __shared__ alignas(32) _Float16 szb[TILE_B * 32];
  __shared__ alignas(32) _Float16 sg[TILE_B * CHUNK];    // tanh output staging (f16)
  __shared__ float sdx[TILE_B * ND];
  __shared__ alignas(16) float scoeff[TILE_B * 96];      // per-step (b, 2c, 3d) staging
  __shared__ alignas(32) _Float16 zbuf[16];              // zero B-fragment half

  const int tid  = threadIdx.x;
  const int lane = tid & 31;
  const int wave = tid >> 5;
  const int ln   = lane & 15;
  const int hi   = lane >> 4;
  const int b0   = blockIdx.x * TILE_B;

  // ---- one-time prep: weights -> f16, fragment-order, into LDS ----
  for (int i = tid; i < 16 * 64; i += 256) {
    int n = i >> 6, k = i & 63;
    sW1t[i] = (n < 15) ? (_Float16)W1[k * 15 + n] : (_Float16)0.0f;
  }
  for (int i = tid; i < 16 * 32; i += 256) {
    int n = i >> 5, k = i & 31;
    _Float16 z = (_Float16)0.0f;
    sW2t[i] = (n < 15 && k < 15) ? (_Float16)W2[k * 15 + n] : z;
    sW3t[i] = (n < 15 && k < 15) ? (_Float16)W3[k * 15 + n] : z;
    sza[i] = z;  // zero-pad once; writers only touch K<16
    szb[i] = z;
  }
  for (int i = tid; i < 128 * 256; i += 256) {  // Wf: [tile][n][k]
    int t = i >> 8, n = (i >> 4) & 15, k = i & 15;
    sWfsw[i] = (k < 15) ? (_Float16)Wf[k * 2048 + t * 16 + n] : (_Float16)0.0f;
  }
  for (int i = tid; i < NF; i += 256) sbf[i] = bf[i];
  if (tid < 16) {
    sb1[tid] = (tid < 15) ? b1[tid] : 0.0f;
    sb2[tid] = (tid < 15) ? b2[tid] : 0.0f;
    sb3[tid] = (tid < 15) ? b3[tid] : 0.0f;
    zbuf[tid] = (_Float16)0.0f;
  }
  // h0 = initial @ W_init + b_init
  for (int i = tid; i < TILE_B * NH; i += 256) {
    int m = i >> 6, h = i & 63;
    float acc = b_init[h];
    const float* ip = initial + (size_t)(b0 + m) * 32;
#pragma unroll
    for (int k = 0; k < 32; ++k) acc += ip[k] * W_init[k * 64 + h];
    sh[i] = acc;
  }
  __syncthreads();

  // ---- sequential scan over 1023 grid intervals ----
  for (int s = 0; s < NSTEP; ++s) {
#ifdef HAVE_TDM
    // TDM: DMA 16 rows x 96 f32 (b,2c,3d) into LDS, row stride 1023*128 elems
    if (tid == 0) {
      unsigned long long ga =
          (unsigned long long)(const void*)(coeffs +
                                            ((size_t)b0 * NSTEP + (size_t)s) * 128 + 32);
      unsigned ldsa = (unsigned)(size_t)(void*)scoeff;
      v4u g0;
      g0[0] = 1u;                                   // count=1 valid descriptor
      g0[1] = ldsa;                                 // lds_addr
      g0[2] = (unsigned)ga;                         // global_addr[31:0]
      g0[3] = (unsigned)((ga >> 32) & 0x01FFFFFFu) | (2u << 30);  // addr hi | type=2
      v8i g1;
      g1[0] = (int)(2u << 16);                      // data_size=4B, no mask/pad
      g1[1] = (int)(96u << 16);                     // tensor_dim0 = 96
      g1[2] = (int)(16u << 16);                     // tensor_dim1 = 16
      g1[3] = (int)(96u << 16);                     // tile_dim0 = 96
      g1[4] = 16;                                   // tile_dim1 = 16 (tile_dim2 = 0)
      g1[5] = 1023 * 128;                           // tensor_dim0_stride (lo 32)
      g1[6] = 0;
      g1[7] = 0;
      v4i z4 = {0, 0, 0, 0};
#if defined(__clang_major__) && (__clang_major__ >= 23)
      v8i z8 = {0, 0, 0, 0, 0, 0, 0, 0};
      __builtin_amdgcn_tensor_load_to_lds(g0, g1, z4, z4, z8, 0);
#else
      __builtin_amdgcn_tensor_load_to_lds(g0, g1, z4, z4, 0);
#endif
      __builtin_amdgcn_s_wait_tensorcnt(0);
    }
#else
    for (int i = tid; i < TILE_B * 96; i += 256) {
      int m = i / 96, d = i % 96;
      const float* p = coeffs + ((size_t)(b0 + m) * NSTEP + s) * 128;
      scoeff[i] = p[32 + d];
      if (d == 0 && s + 1 < NSTEP) {
        __builtin_prefetch(p + 128 + 32, 0, 0);
        __builtin_prefetch(p + 128 + 96, 0, 0);
      }
    }
#endif
    __syncthreads();

    // 4 RK4 (3/8-rule) evaluations
    for (int e = 0; e < 4; ++e) {
      const float frac = (e == 0) ? 0.0f
                       : (e == 1) ? (1.0f / 3.0f)
                       : (e == 2) ? (2.0f / 3.0f)
                       : 1.0f;
      for (int i = tid; i < TILE_B * NH; i += 256) {
        float hv = sh[i];
        if (e == 1)      hv += sk1[i] * (1.0f / 3.0f);
        else if (e == 2) hv += sk2[i] - sk1[i] * (1.0f / 3.0f);
        else if (e == 3) hv += sk1[i] - sk2[i] + sk3[i];
        shevh[i] = (_Float16)hv;
      }
      for (int i = tid; i < TILE_B * ND; i += 256) {
        int m = i >> 5, d = i & 31;
        const float* cp = &scoeff[m * 96 + d];
        sdx[i] = cp[0] + (cp[32] + cp[64] * frac) * frac;
      }
      __syncthreads();

      // ---- small GEMM chain (wave 0; EXEC all-ones within the wave) ----
      if (wave == 0) {
        v16h a0 = loadA(&shevh[ln * NH], hi);
        v16h a1 = loadA(&shevh[ln * NH + 32], hi);
        v16h bW0 = *(const v16h*)&sW1t[ln * 64 + hi * 16];
        v16h bW1 = *(const v16h*)&sW1t[ln * 64 + 32 + hi * 16];
        v8f c = {};
        c = wmma16(a0, bW0, c);
        c = wmma16(a1, bW1, c);
#pragma unroll
        for (int r = 0; r < 8; ++r)
          sza[(r + hi * 8) * 32 + ln] = (_Float16)fmaxf(c[r] + sb1[ln], 0.0f);

        v16h a2 = loadA(&sza[ln * 32], hi);
        v16h bW2 = *(const v16h*)&sW2t[ln * 32 + hi * 16];
        v8f c2 = {};
        c2 = wmma16(a2, bW2, c2);
#pragma unroll
        for (int r = 0; r < 8; ++r)
          szb[(r + hi * 8) * 32 + ln] = (_Float16)fmaxf(c2[r] + sb2[ln], 0.0f);

        v16h a3 = loadA(&szb[ln * 32], hi);
        v16h bW3 = *(const v16h*)&sW3t[ln * 32 + hi * 16];
        v8f c3 = {};
        c3 = wmma16(a3, bW3, c3);
#pragma unroll
        for (int r = 0; r < 8; ++r)
          sza[(r + hi * 8) * 32 + ln] = (_Float16)fmaxf(c3[r] + sb3[ln], 0.0f);
      }
      __syncthreads();

      // ---- big GEMM z3 @ Wf (8 waves x 16 N-tiles) + tanh + dx contraction ----
      v16h aF = loadA(&sza[ln * 32], hi);

      for (int ch = 0; ch < 2; ++ch) {
#pragma unroll
        for (int t = 0; t < 8; ++t) {
          int tile = ch * 64 + wave * 8 + t;  // 0..127
          const _Float16* bp = hi ? zbuf : &sWfsw[tile * 256 + ln * 16];
          v16h bF = *(const v16h*)bp;
          v8f c = {};
          c = wmma16(aF, bF, c);
          int ng = tile * 16 + ln;
          int nl = ng & (CHUNK - 1);
#pragma unroll
          for (int r = 0; r < 8; ++r) {
            float v = dev_tanh(c[r] + sbf[ng]);
            sg[(r + hi * 8) * CHUNK + nl] = (_Float16)v;
          }
        }
        __syncthreads();
        // deterministic reduction: k[m,h] = sum_d g[m, h*32+d] * dx[m,d]
        for (int o = tid; o < TILE_B * 32; o += 256) {
          int m = o >> 5, hl = o & 31;
          const _Float16* gp = &sg[m * CHUNK + hl * 32];
          const float* dxp = &sdx[m * ND];
          v16h g0 = *(const v16h*)gp;
          v16h g1 = *(const v16h*)(gp + 16);
          float acc = 0.0f;
#pragma unroll
          for (int d = 0; d < 16; ++d) acc += (float)g0[d] * dxp[d];
#pragma unroll
          for (int d = 0; d < 16; ++d) acc += (float)g1[d] * dxp[16 + d];
          skc[m * NH + ch * 32 + hl] = acc;
        }
        __syncthreads();
      }

      // commit k_e / advance h
      for (int i = tid; i < TILE_B * NH; i += 256) {
        float kv = skc[i];
        if (e == 0)      sk1[i] = kv;
        else if (e == 1) sk2[i] = kv;
        else if (e == 2) sk3[i] = kv;
        else             sh[i] += (sk1[i] + 3.0f * (sk2[i] + sk3[i]) + kv) * 0.125f;
      }
      __syncthreads();
    }
  }

  // ---- readout: out = hT @ W_out + b_out ----
  for (int o = tid; o < TILE_B * 10; o += 256) {
    int m = o / 10, j = o % 10;
    float acc = b_out[j];
#pragma unroll
    for (int h = 0; h < NH; ++h) acc += sh[m * NH + h] * W_out[h * 10 + j];
    out[(size_t)(b0 + m) * 10 + j] = acc;
  }
}

extern "C" void kernel_launch(void* const* d_in, const int* in_sizes, int n_in,
                              void* d_out, int out_size, void* d_ws, size_t ws_size,
                              hipStream_t stream) {
  (void)in_sizes; (void)n_in; (void)d_ws; (void)ws_size; (void)out_size;
  const float* coeffs  = (const float*)d_in[0];
  const float* initial = (const float*)d_in[1];
  const float* W_init  = (const float*)d_in[2];
  const float* b_init  = (const float*)d_in[3];
  const float* W1      = (const float*)d_in[4];
  const float* b1      = (const float*)d_in[5];
  const float* W2      = (const float*)d_in[6];
  const float* b2      = (const float*)d_in[7];
  const float* W3      = (const float*)d_in[8];
  const float* b3      = (const float*)d_in[9];
  const float* Wf      = (const float*)d_in[10];
  const float* bf      = (const float*)d_in[11];
  const float* W_out   = (const float*)d_in[12];
  const float* b_out   = (const float*)d_in[13];
  float* out = (float*)d_out;

  neural_cde_kernel<<<dim3(256 / TILE_B), dim3(256), 0, stream>>>(
      coeffs, initial, W_init, b_init, W1, b1, W2, b2, W3, b3, Wf, bf,
      W_out, b_out, out);
}
